// update_e_v2_noself_1245540515927
// MI455X (gfx1250) — matrix operand order repaired and checked by reading, not compiled, not run on hardware
//
#include <hip/hip_runtime.h>
#include <hip/hip_bf16.h>

// ---------------------------------------------------------------------------
// DimeNet++ interaction block for MI455X (gfx1250, wave32, WMMA).
// Edge GEMMs: f16-in / f32-acc v_wmma_f32_16x16x32_f16 on 32-row panels
// (each B weight fragment feeds 2 WMMAs). Fused 8-stage residual chain in
// double-buffered LDS. Triplet side: bandwidth-optimized gather / atomic
// segment-sum with f16 gather payload. silu via v_rcp_f32 (no IEEE div).
// ---------------------------------------------------------------------------

typedef __attribute__((ext_vector_type(16))) _Float16 v16h;
typedef __attribute__((ext_vector_type(8)))  _Float16 v8h;
typedef __attribute__((ext_vector_type(8)))  float    v8f;

#define H 128
#define ICH 64
#define NR 6
#define NSR 42
#define ROWS 32   // edges per block (two 16-row WMMA panels)
#define LDA 144   // LDS stride (halves) for 128-wide panels (16B-aligned rows, skewed)
#define LDX 80    // LDS stride (halves) for 64-wide panels

__device__ __forceinline__ float silu_f(float x) {
    // x * rcp(1 + exp(-x)) : v_exp_f32 + v_add + v_rcp_f32 + v_mul
    return x * __builtin_amdgcn_rcpf(1.0f + __expf(-x));
}

// A fragment (16x32 f16, MxK): lane L holds row M=L&15.
// lanes 0-15: K = k0+0..7 (vgpr0-3), k0+16..23 (vgpr4-7)
// lanes 16-31: K = k0+8..15,         k0+24..31
__device__ __forceinline__ v16h load_a_frag(const _Float16* s, int stride, int k0, int lane) {
    int row = lane & 15;
    int kb  = k0 + ((lane & 16) ? 8 : 0);
    const _Float16* p = s + row * stride + kb;
    v8h lo = *reinterpret_cast<const v8h*>(p);        // ds_load_b128
    v8h hi = *reinterpret_cast<const v8h*>(p + 16);   // ds_load_b128
    return __builtin_shufflevector(lo, hi, 0,1,2,3,4,5,6,7,8,9,10,11,12,13,14,15);
}

// B fragment (32x16 f16, KxN) for act @ W^T: B[k][n] = W[n][k], W row-major f16.
// lane L holds col N=n0+(L&15); lanes 0-15: K=k0+0..7,16..23, lanes 16-31: K=k0+8..15,24..31.
__device__ __forceinline__ v16h load_b_frag(const _Float16* W, int indim, int n0, int k0, int lane) {
    int n  = n0 + (lane & 15);
    int kb = k0 + ((lane & 16) ? 8 : 0);
    const _Float16* p = W + (size_t)n * indim + kb;
    v8h lo = *reinterpret_cast<const v8h*>(p);        // global_load_b128
    v8h hi = *reinterpret_cast<const v8h*>(p + 16);   // global_load_b128
    return __builtin_shufflevector(lo, hi, 0,1,2,3,4,5,6,7,8,9,10,11,12,13,14,15);
}

// Two 16x16 output tiles (rows 0-15 and 16-31) of src(32xK) @ W^T, K = 32*nk.
// Each B fragment is reused for both row panels.
__device__ __forceinline__ void wmma_panel2(const _Float16* src, int sstride,
                                            const _Float16* W, int indim,
                                            int n0, int lane, int nk,
                                            v8f& c0, v8f& c1) {
    for (int kc = 0; kc < nk; ++kc) {
        v16h b  = load_b_frag(W, indim, n0, kc * 32, lane);
        v16h a0 = load_a_frag(src, sstride, kc * 32, lane);
        v16h a1 = load_a_frag(src + 16 * sstride, sstride, kc * 32, lane);
        c0 = __builtin_amdgcn_wmma_f32_16x16x32_f16(false, a0, false, b, (short)0, c0, false, false);
        c1 = __builtin_amdgcn_wmma_f32_16x16x32_f16(false, a1, false, b, (short)0, c1, false, false);
    }
}

__device__ __forceinline__ void bias_silu2(v8f& c0, v8f& c1, const float* bias, int n0, int lane) {
    float bv = bias[n0 + (lane & 15)];
#pragma unroll
    for (int r = 0; r < 8; ++r) {
        c0[r] = silu_f(c0[r] + bv);
        c1[r] = silu_f(c1[r] + bv);
    }
}

__device__ __forceinline__ void silu2(v8f& c0, v8f& c1) {
#pragma unroll
    for (int r = 0; r < 8; ++r) {
        c0[r] = silu_f(c0[r]);
        c1[r] = silu_f(c1[r]);
    }
}

// Store 16x16 f32 tile (WMMA C layout) as f16 into LDS panel.
__device__ __forceinline__ void store_tile(const v8f& c, _Float16* dst, int dstride, int n0, int lane) {
    int n  = n0 + (lane & 15);
    int mb = (lane & 16) ? 8 : 0;
#pragma unroll
    for (int r = 0; r < 8; ++r) dst[(mb + r) * dstride + n] = (_Float16)c[r];
}

__device__ __forceinline__ void store_tile2(const v8f& c0, const v8f& c1, _Float16* dst, int n0, int lane) {
    store_tile(c0, dst, LDA, n0, lane);
    store_tile(c1, dst + 16 * LDA, LDA, n0, lane);
}

// ---------------------------------------------------------------------------
__global__ void k_f32_to_f16(const float* __restrict__ src, _Float16* __restrict__ dst, int n) {
    int i = blockIdx.x * 256 + threadIdx.x;
    if (i < n) dst[i] = (_Float16)src[i];
}

// W_rbf_comb[h][r] = sum_b W_rbf2[h][b] * W_rbf1[b][r], padded to [128][8]
__global__ void k_combo_rbf(const float* __restrict__ W1, const float* __restrict__ W2,
                            float* __restrict__ out) {
    int i = blockIdx.x * 256 + threadIdx.x;
    if (i < H * 8) {
        int h = i >> 3, r = i & 7;
        float a = 0.f;
        if (r < NR) {
#pragma unroll
            for (int b = 0; b < 8; ++b) a += W2[h * 8 + b] * W1[b * NR + r];
        }
        out[i] = a;
    }
}

// ---------------------------------------------------------------------------
// x_down = silu( (silu(x1 @ Wkj^T + b) * rbf) @ Wdown^T )   (f16 out, [E][64])
__global__ __launch_bounds__(256) void k_edge_front(
    const float* __restrict__ x1, const float* __restrict__ rbf0,
    const _Float16* __restrict__ Wkj16, const float* __restrict__ b_kj,
    const _Float16* __restrict__ Wdown16, const float* __restrict__ Wrbfc,
    _Float16* __restrict__ xdown, int E) {
    __shared__ _Float16 sA[ROWS * LDA];
    __shared__ _Float16 sAct[ROWS * LDA];
    __shared__ float    sR[ROWS * H];
    __shared__ float    sRbf0[ROWS * 8];
    __shared__ float    sWr[H * 8];

    int e0 = blockIdx.x * ROWS;
    int tid = threadIdx.x;

    if (tid < ROWS * 6) {
        int m = tid / 6, r = tid % 6;
        int e = e0 + m; if (e >= E) e = E - 1;
        sRbf0[m * 8 + r] = rbf0[(size_t)e * NR + r];
    }
    for (int i = tid; i < H * 8; i += 256) sWr[i] = Wrbfc[i];
    for (int i = tid; i < ROWS * H; i += 256) {
        int m = i >> 7, n = i & 127;
        int e = e0 + m; if (e >= E) e = E - 1;
        sA[m * LDA + n] = (_Float16)x1[(size_t)e * H + n];
    }
    __syncthreads();

    for (int i = tid; i < ROWS * H; i += 256) {
        int m = i >> 7, h = i & 127;
        float acc = 0.f;
#pragma unroll
        for (int r = 0; r < NR; ++r) acc += sRbf0[m * 8 + r] * sWr[h * 8 + r];
        sR[i] = acc;
    }
    __syncthreads();

    int wave = tid >> 5, lane = tid & 31;

    // GEMM 1: x1 @ Wkj^T  (each wave: two 16x16 tiles sharing B, K=128)
    v8f c0 = {}, c1 = {};
    wmma_panel2(sA, LDA, Wkj16, H, wave * 16, lane, 4, c0, c1);
    {
        int n  = wave * 16 + (lane & 15);
        int mb = (lane & 16) ? 8 : 0;
        float bv = b_kj[n];
#pragma unroll
        for (int r = 0; r < 8; ++r) {
            int m0 = mb + r, m1 = 16 + mb + r;
            sAct[m0 * LDA + n] = (_Float16)(silu_f(c0[r] + bv) * sR[m0 * H + n]);
            sAct[m1 * LDA + n] = (_Float16)(silu_f(c1[r] + bv) * sR[m1 * H + n]);
        }
    }
    __syncthreads();

    // GEMM 2: down-projection to 64 channels (waves 0..3)
    if (wave < 4) {
        v8f d0 = {}, d1 = {};
        wmma_panel2(sAct, LDA, Wdown16, H, wave * 16, lane, 4, d0, d1);
        int n  = wave * 16 + (lane & 15);
        int mb = (lane & 16) ? 8 : 0;
#pragma unroll
        for (int r = 0; r < 8; ++r) {
            int ea = e0 + mb + r, eb = e0 + 16 + mb + r;
            if (ea < E) xdown[(size_t)ea * ICH + n] = (_Float16)silu_f(d0[r]);
            if (eb < E) xdown[(size_t)eb * ICH + n] = (_Float16)silu_f(d1[r]);
        }
    }
}

// ---------------------------------------------------------------------------
// p[t][b] = sbf[t][:] . W_sbf1[b][:]   (bottleneck projection 42 -> 8)
__global__ __launch_bounds__(256) void k_sbf1(const float* __restrict__ sbf,
                                              const float* __restrict__ Wsbf1,
                                              float* __restrict__ p, long T) {
    __shared__ float sW[8 * NSR];
    for (int i = threadIdx.x; i < 8 * NSR; i += 256) sW[i] = Wsbf1[i];
    __syncthreads();
    long nwork = T * 8;
    for (long i = (long)blockIdx.x * 256 + threadIdx.x; i < nwork; i += (long)gridDim.x * 256) {
        long t = i >> 3; int b = (int)(i & 7);
        const float* row = sbf + t * NSR;
        float acc = 0.f;
#pragma unroll
        for (int j = 0; j < NSR; ++j) acc += row[j] * sW[b * NSR + j];
        p[t * 8 + b] = acc;
    }
}

// ---------------------------------------------------------------------------
// Triplet gather + scale + segment-sum scatter. One wave per triplet, 2 ch/lane.
__global__ __launch_bounds__(256) void k_triplet(
    const float* __restrict__ p, const int* __restrict__ idx_kj,
    const int* __restrict__ idx_ji, const float* __restrict__ Wsbf2,
    const _Float16* __restrict__ xdown, float* __restrict__ xagg,
    long T, long nwaves) {
    __shared__ float sW[ICH * 8];
    for (int i = threadIdx.x; i < ICH * 8; i += 256) sW[i] = Wsbf2[i];
    __syncthreads();
    int lane = threadIdx.x & 31;
    int c0 = lane * 2, c1 = lane * 2 + 1;
    long w = (long)blockIdx.x * 8 + (threadIdx.x >> 5);
    for (long t = w; t < T; t += nwaves) {
        int kj = idx_kj[t], ji = idx_ji[t];
        const float* pr = p + t * 8;
        float s0 = 0.f, s1 = 0.f;
#pragma unroll
        for (int b = 0; b < 8; ++b) {
            float pv = pr[b];
            s0 += pv * sW[c0 * 8 + b];
            s1 += pv * sW[c1 * 8 + b];
        }
        const _Float16* xd = xdown + (size_t)kj * ICH + c0;   // coalesced f16 gather
        float v0 = (float)xd[0] * s0;
        float v1 = (float)xd[1] * s1;
        float* dst = xagg + (size_t)ji * ICH + c0;            // coalesced f32 atomics
        atomicAdd(dst, v0);
        atomicAdd(dst + 1, v1);
    }
}

// ---------------------------------------------------------------------------
// Fused back-end: e1 = resid chain(silu(xagg @ Wup^T)) ; e2 = (rbf0 @ Wrbf^T)*e1
// Eight GEMM stages over a 32x128 panel, double-buffered in LDS.
__global__ __launch_bounds__(256) void k_edge_back(
    const float* __restrict__ xagg, const float* __restrict__ rbf0,
    const _Float16* __restrict__ Wup16,
    const _Float16* __restrict__ Wb1a16, const float* __restrict__ bb1a,
    const _Float16* __restrict__ Wb1b16, const float* __restrict__ bb1b,
    const _Float16* __restrict__ Wlin16, const float* __restrict__ b_lin,
    const _Float16* __restrict__ Wa1a16, const float* __restrict__ ba1a,
    const _Float16* __restrict__ Wa1b16, const float* __restrict__ ba1b,
    const _Float16* __restrict__ Wa2a16, const float* __restrict__ ba2a,
    const _Float16* __restrict__ Wa2b16, const float* __restrict__ ba2b,
    const float* __restrict__ W_rbf, float* __restrict__ out, int E) {
    __shared__ _Float16 sX[ROWS * LDX];
    __shared__ _Float16 sBufA[ROWS * LDA];
    __shared__ _Float16 sBufB[ROWS * LDA];
    __shared__ float    sRbf0[ROWS * 8];

    int e0 = blockIdx.x * ROWS;
    int tid = threadIdx.x;
    int wave = tid >> 5, lane = tid & 31;
    int n0 = wave * 16;

    for (int i = tid; i < ROWS * ICH; i += 256) {
        int m = i >> 6, n = i & 63;
        int e = e0 + m; if (e >= E) e = E - 1;
        sX[m * LDX + n] = (_Float16)xagg[(size_t)e * ICH + n];
    }
    if (tid < ROWS * 6) {
        int m = tid / 6, r = tid % 6;
        int e = e0 + m; if (e >= E) e = E - 1;
        sRbf0[m * 8 + r] = rbf0[(size_t)e * NR + r];
    }
    __syncthreads();

    // P0: c_run = silu(xagg @ Wup^T)               -> write A
    v8f cr0 = {}, cr1 = {};
    wmma_panel2(sX, LDX, Wup16, ICH, n0, lane, 2, cr0, cr1);
    silu2(cr0, cr1);
    store_tile2(cr0, cr1, sBufA, n0, lane);
    __syncthreads();

    // P1: t = silu(A @ Wb1a^T + bb1a)              -> write B
    v8f t0 = {}, t1 = {};
    wmma_panel2(sBufA, LDA, Wb1a16, H, n0, lane, 4, t0, t1);
    bias_silu2(t0, t1, bb1a, n0, lane);
    store_tile2(t0, t1, sBufB, n0, lane);
    __syncthreads();

    // P2: c_run += silu(B @ Wb1b^T + bb1b)         -> write A
    t0 = (v8f){}; t1 = (v8f){};
    wmma_panel2(sBufB, LDA, Wb1b16, H, n0, lane, 4, t0, t1);
    bias_silu2(t0, t1, bb1b, n0, lane);
    cr0 = cr0 + t0; cr1 = cr1 + t1;
    store_tile2(cr0, cr1, sBufA, n0, lane);
    __syncthreads();

    // P3: c_run = silu(A @ Wlin^T + b_lin)         -> write B
    cr0 = (v8f){}; cr1 = (v8f){};
    wmma_panel2(sBufA, LDA, Wlin16, H, n0, lane, 4, cr0, cr1);
    bias_silu2(cr0, cr1, b_lin, n0, lane);
    store_tile2(cr0, cr1, sBufB, n0, lane);
    __syncthreads();

    // P4: t = silu(B @ Wa1a^T + ba1a)              -> write A
    t0 = (v8f){}; t1 = (v8f){};
    wmma_panel2(sBufB, LDA, Wa1a16, H, n0, lane, 4, t0, t1);
    bias_silu2(t0, t1, ba1a, n0, lane);
    store_tile2(t0, t1, sBufA, n0, lane);
    __syncthreads();

    // P5: c_run += silu(A @ Wa1b^T + ba1b)         -> write B
    t0 = (v8f){}; t1 = (v8f){};
    wmma_panel2(sBufA, LDA, Wa1b16, H, n0, lane, 4, t0, t1);
    bias_silu2(t0, t1, ba1b, n0, lane);
    cr0 = cr0 + t0; cr1 = cr1 + t1;
    store_tile2(cr0, cr1, sBufB, n0, lane);
    __syncthreads();

    // P6: t = silu(B @ Wa2a^T + ba2a)              -> write A
    t0 = (v8f){}; t1 = (v8f){};
    wmma_panel2(sBufB, LDA, Wa2a16, H, n0, lane, 4, t0, t1);
    bias_silu2(t0, t1, ba2a, n0, lane);
    store_tile2(t0, t1, sBufA, n0, lane);
    __syncthreads();

    // P7: c_run += silu(A @ Wa2b^T + ba2b)   (final e1)
    t0 = (v8f){}; t1 = (v8f){};
    wmma_panel2(sBufA, LDA, Wa2b16, H, n0, lane, 4, t0, t1);
    bias_silu2(t0, t1, ba2b, n0, lane);
    cr0 = cr0 + t0; cr1 = cr1 + t1;

    // Output e1 and e2 = (rbf0 @ W_rbf^T) * e1
    int n  = n0 + (lane & 15);
    int mb = (lane & 16) ? 8 : 0;
    float wr[NR];
#pragma unroll
    for (int j = 0; j < NR; ++j) wr[j] = W_rbf[n * NR + j];
    size_t e2base = (size_t)E * H;
#pragma unroll
    for (int r = 0; r < 8; ++r) {
        int m0 = mb + r, m1 = 16 + mb + r;
        int ea = e0 + m0, eb = e0 + m1;
        if (ea < E) {
            float e1v = cr0[r];
            out[(size_t)ea * H + n] = e1v;
            float acc = 0.f;
#pragma unroll
            for (int j = 0; j < NR; ++j) acc += sRbf0[m0 * 8 + j] * wr[j];
            out[e2base + (size_t)ea * H + n] = acc * e1v;
        }
        if (eb < E) {
            float e1v = cr1[r];
            out[(size_t)eb * H + n] = e1v;
            float acc = 0.f;
#pragma unroll
            for (int j = 0; j < NR; ++j) acc += sRbf0[m1 * 8 + j] * wr[j];
            out[e2base + (size_t)eb * H + n] = acc * e1v;
        }
    }
}

// ---------------------------------------------------------------------------
extern "C" void kernel_launch(void* const* d_in, const int* in_sizes, int n_in,
                              void* d_out, int out_size, void* d_ws, size_t ws_size,
                              hipStream_t stream) {
    const float* x1     = (const float*)d_in[0];
    const float* rbf0   = (const float*)d_in[1];
    const float* sbf    = (const float*)d_in[2];
    const int*   idx_kj = (const int*)d_in[3];
    const int*   idx_ji = (const int*)d_in[4];
    const float* W_rbf1 = (const float*)d_in[5];
    const float* W_rbf2 = (const float*)d_in[6];
    const float* W_sbf1 = (const float*)d_in[7];
    const float* W_sbf2 = (const float*)d_in[8];
    const float* W_rbf  = (const float*)d_in[9];
    const float* W_kj   = (const float*)d_in[10];
    const float* b_kj   = (const float*)d_in[11];
    const float* W_down = (const float*)d_in[12];
    const float* W_up   = (const float*)d_in[13];
    const float* Wb1a = (const float*)d_in[14]; const float* bb1a = (const float*)d_in[15];
    const float* Wb1b = (const float*)d_in[16]; const float* bb1b = (const float*)d_in[17];
    const float* W_lin = (const float*)d_in[18]; const float* b_lin = (const float*)d_in[19];
    const float* Wa1a = (const float*)d_in[20]; const float* ba1a = (const float*)d_in[21];
    const float* Wa1b = (const float*)d_in[22]; const float* ba1b = (const float*)d_in[23];
    const float* Wa2a = (const float*)d_in[24]; const float* ba2a = (const float*)d_in[25];
    const float* Wa2b = (const float*)d_in[26]; const float* ba2b = (const float*)d_in[27];

    int  E = in_sizes[0] / H;          // 400000
    long T = (long)in_sizes[2] / NSR;  // 2000000

    // Workspace carve-up
    char* ws = (char*)d_ws;
    size_t off = 0;
    auto carve = [&](size_t bytes) -> void* {
        void* r = ws + off;
        off += (bytes + 255) & ~(size_t)255;
        return r;
    };
    _Float16* Wkj16   = (_Float16*)carve(H * H * 2);
    _Float16* Wdown16 = (_Float16*)carve(ICH * H * 2);
    _Float16* Wup16   = (_Float16*)carve(H * ICH * 2);
    _Float16* Wb1a16  = (_Float16*)carve(H * H * 2);
    _Float16* Wb1b16  = (_Float16*)carve(H * H * 2);
    _Float16* Wlin16  = (_Float16*)carve(H * H * 2);
    _Float16* Wa1a16  = (_Float16*)carve(H * H * 2);
    _Float16* Wa1b16  = (_Float16*)carve(H * H * 2);
    _Float16* Wa2a16  = (_Float16*)carve(H * H * 2);
    _Float16* Wa2b16  = (_Float16*)carve(H * H * 2);
    float*    Wrbfc   = (float*)carve(H * 8 * 4);
    _Float16* xdown   = (_Float16*)carve((size_t)E * ICH * 2);
    float*    xagg    = (float*)carve((size_t)E * ICH * 4);
    float*    psbf    = (float*)carve((size_t)T * 8 * 4);
    (void)ws_size; (void)n_in; (void)out_size;

    // Segment-sum accumulator must be zero each call.
    hipMemsetAsync(xagg, 0, (size_t)E * ICH * sizeof(float), stream);

    // Weight preparation (tiny)
    k_f32_to_f16<<<(H * H + 255) / 256, 256, 0, stream>>>(W_kj, Wkj16, H * H);
    k_f32_to_f16<<<(ICH * H + 255) / 256, 256, 0, stream>>>(W_down, Wdown16, ICH * H);
    k_f32_to_f16<<<(H * ICH + 255) / 256, 256, 0, stream>>>(W_up, Wup16, H * ICH);
    k_f32_to_f16<<<(H * H + 255) / 256, 256, 0, stream>>>(Wb1a, Wb1a16, H * H);
    k_f32_to_f16<<<(H * H + 255) / 256, 256, 0, stream>>>(Wb1b, Wb1b16, H * H);
    k_f32_to_f16<<<(H * H + 255) / 256, 256, 0, stream>>>(W_lin, Wlin16, H * H);
    k_f32_to_f16<<<(H * H + 255) / 256, 256, 0, stream>>>(Wa1a, Wa1a16, H * H);
    k_f32_to_f16<<<(H * H + 255) / 256, 256, 0, stream>>>(Wa1b, Wa1b16, H * H);
    k_f32_to_f16<<<(H * H + 255) / 256, 256, 0, stream>>>(Wa2a, Wa2a16, H * H);
    k_f32_to_f16<<<(H * H + 255) / 256, 256, 0, stream>>>(Wa2b, Wa2b16, H * H);
    k_combo_rbf<<<(H * 8 + 255) / 256, 256, 0, stream>>>(W_rbf1, W_rbf2, Wrbfc);

    int ntiles = (E + ROWS - 1) / ROWS;

    // Edge front: x_down = silu((silu(x1 Wkj^T+b) * rbf) Wdown^T)
    k_edge_front<<<ntiles, 256, 0, stream>>>(x1, rbf0, Wkj16, b_kj, Wdown16, Wrbfc, xdown, E);

    // sbf bottleneck projection (42 -> 8)
    k_sbf1<<<16384, 256, 0, stream>>>(sbf, W_sbf1, psbf, T);

    // Triplet gather/scatter segment sum
    const long TB = 4096;
    k_triplet<<<(int)TB, 256, 0, stream>>>(psbf, idx_kj, idx_ji, W_sbf2, xdown, xagg, T, TB * 8);

    // Fused back-end: W_up + residual chain + e2
    k_edge_back<<<ntiles, 256, 0, stream>>>(
        xagg, rbf0, Wup16,
        Wb1a16, bb1a, Wb1b16, bb1b, Wlin16, b_lin,
        Wa1a16, ba1a, Wa1b16, ba1b, Wa2a16, ba2a, Wa2b16, ba2b,
        W_rbf, (float*)d_out, E);
}
